// Block_57140244906730
// MI455X (gfx1250) — matrix operand order repaired and checked by reading, not compiled, not run on hardware
//
#include <hip/hip_runtime.h>
#include <hip/hip_bf16.h>
#include <math.h>

// ---------------------------------------------------------------------------
// Types & helpers
// ---------------------------------------------------------------------------
typedef __attribute__((ext_vector_type(16))) __bf16 v16bf;
typedef __attribute__((ext_vector_type(8)))  float  v8f;
typedef int v4i __attribute__((vector_size(16)));   // matches async builtin param

union V16 { v16bf v; unsigned short u[16]; };

__device__ __forceinline__ unsigned short f2b(float f) {
  unsigned u = __float_as_uint(f);
  u += 0x7FFFu + ((u >> 16) & 1u);          // round-to-nearest-even
  return (unsigned short)(u >> 16);
}

// 16-bit A-matrix 16x32 layout (ISA 7.12.2): lane<16 -> K 0-7 / 16-23,
// lane>=16 -> K 8-15 / 24-31. idx = 2*vgpr + half.
__device__ __forceinline__ int a_frag_k(int idx, int kbase) {
  int v = idx >> 1;
  return ((v & 4) ? 16 : 0) + kbase + ((v & 3) << 1) + (idx & 1);
}

__device__ __forceinline__ v8f wmma_bf16(v16bf a, v16bf b, v8f c) {
  return __builtin_amdgcn_wmma_f32_16x16x32_bf16(false, a, false, b,
                                                 (short)0, c, false, false);
}

// --- CDNA5 async global->LDS copy (ASYNCcnt) with portable fallback --------
#if __has_builtin(__builtin_amdgcn_global_load_async_to_lds_b128) && \
    __has_builtin(__builtin_amdgcn_s_wait_asynccnt)
#define HAVE_ASYNC_LDS 1
#else
#define HAVE_ASYNC_LDS 0
#endif

__device__ __forceinline__ void cp_async_b128(void* lds, const void* g) {
#if HAVE_ASYNC_LDS
  __builtin_amdgcn_global_load_async_to_lds_b128(
      (__attribute__((address_space(1))) v4i*)(void*)g,
      (__attribute__((address_space(3))) v4i*)lds, 0, 0);
#else
  *(uint4*)lds = *(const uint4*)g;
#endif
}

__device__ __forceinline__ void cp_async_wait() {
#if HAVE_ASYNC_LDS
  __builtin_amdgcn_s_wait_asynccnt(0);
#endif
}

// ---------------------------------------------------------------------------
// Generic bf16 WMMA GEMM: C[M,N] = A[M,K] * B[K,N] (+bias[N]) (+addend[M,N])
// A,B row-major bf16 (ushort bits), C f32 row-major.
// Double-buffered LDS staging via async-to-LDS B128 copies.
// ---------------------------------------------------------------------------
#define BM 128
#define BN 128
#define BK 32
#define LDA_S 40     // ushorts; 80B rows -> every 8-elem chunk 16B aligned
#define LDB_S 136    // ushorts; 272B rows

__device__ __forceinline__ void stage_tile_a(
    unsigned short* __restrict__ dst, const unsigned short* __restrict__ A,
    int lda, int bm, int k0, int M, int K, int tid)
{
  if (bm + BM <= M && k0 + BK <= K && (lda & 7) == 0) {
    for (int c = tid; c < (BM * BK) / 8; c += 256) {   // 512 chunks of 8
      int r = c >> 2, col = (c & 3) * 8;
      cp_async_b128(&dst[r * LDA_S + col], &A[(size_t)(bm + r) * lda + k0 + col]);
    }
  } else {
    for (int i = tid; i < BM * BK; i += 256) {
      int r = i >> 5, c = i & 31;
      int gr = bm + r, gk = k0 + c;
      dst[r * LDA_S + c] = (gr < M && gk < K) ? A[(size_t)gr * lda + gk]
                                              : (unsigned short)0;
    }
  }
}

__device__ __forceinline__ void stage_tile_b(
    unsigned short* __restrict__ dst, const unsigned short* __restrict__ B,
    int ldb, int bn, int k0, int N, int K, int tid)
{
  if (bn + BN <= N && k0 + BK <= K && (ldb & 7) == 0) {
    for (int c = tid; c < (BK * BN) / 8; c += 256) {   // 512 chunks of 8
      int r = c >> 4, col = (c & 15) * 8;
      cp_async_b128(&dst[r * LDB_S + col], &B[(size_t)(k0 + r) * ldb + bn + col]);
    }
  } else {
    for (int i = tid; i < BK * BN; i += 256) {
      int r = i >> 7, c = i & 127;
      int gk = k0 + r, gc = bn + c;
      dst[r * LDB_S + c] = (gk < K && gc < N) ? B[(size_t)gk * ldb + gc]
                                              : (unsigned short)0;
    }
  }
}

__global__ __launch_bounds__(256) void gemm_bf16_kernel(
    const unsigned short* __restrict__ A, int lda,
    const unsigned short* __restrict__ B, int ldb,
    float* __restrict__ C, int ldc,
    int M, int N, int K,
    const float* __restrict__ bias,
    const float* __restrict__ addend)
{
  __shared__ __align__(16) unsigned short As[2][BM * LDA_S];
  __shared__ __align__(16) unsigned short Bs[2][BK * LDB_S];

  const int tid  = threadIdx.x;
  const int lane = tid & 31;
  const int wid  = tid >> 5;
  const int wm   = wid >> 2;      // 0..1 -> 64 rows each
  const int wn   = wid & 3;       // 0..3 -> 32 cols each
  const int lrow = lane & 15;
  const int lhalf = lane >> 4;
  const int bm = blockIdx.y * BM;
  const int bn = blockIdx.x * BN;

  v8f acc[4][2] = {};

  // prologue: stage tile 0
  stage_tile_a(As[0], A, lda, bm, 0, M, K, tid);
  stage_tile_b(Bs[0], B, ldb, bn, 0, N, K, tid);
  cp_async_wait();
  __syncthreads();

  int cur = 0;
  for (int k0 = 0; k0 < K; k0 += BK, cur ^= 1) {
    // overlap: stage next tile while computing on current
    int kn = k0 + BK;
    if (kn < K) {
      stage_tile_a(As[cur ^ 1], A, lda, bm, kn, M, K, tid);
      stage_tile_b(Bs[cur ^ 1], B, ldb, bn, kn, N, K, tid);
    }

    const unsigned short* __restrict__ Asc = As[cur];
    const unsigned short* __restrict__ Bsc = Bs[cur];

    // B fragments (32x16 each): lane = column, k across halfs
    v16bf bf[2];
#pragma unroll
    for (int ni = 0; ni < 2; ++ni) {
      V16 t;
      int n  = wn * 32 + ni * 16 + lrow;
      int kb = lhalf * 8;
#pragma unroll
      for (int idx = 0; idx < 16; ++idx)
        t.u[idx] = Bsc[a_frag_k(idx, kb) * LDB_S + n];
      bf[ni] = t.v;
    }
#pragma unroll
    for (int mi = 0; mi < 4; ++mi) {
      V16 t;
      int row = wm * 64 + mi * 16 + lrow;
      int kb  = lhalf * 8;
#pragma unroll
      for (int idx = 0; idx < 16; ++idx)
        t.u[idx] = Asc[row * LDA_S + a_frag_k(idx, kb)];
      acc[mi][0] = wmma_bf16(t.v, bf[0], acc[mi][0]);
      acc[mi][1] = wmma_bf16(t.v, bf[1], acc[mi][1]);
    }

    cp_async_wait();     // our async copies into the other buffer are done
    __syncthreads();     // everyone finished reading cur & staging next
  }

  // epilogue: C layout VGPR r -> row r (lanes0-15) / r+8 (lanes16-31)
#pragma unroll
  for (int mi = 0; mi < 4; ++mi)
#pragma unroll
    for (int ni = 0; ni < 2; ++ni)
#pragma unroll
      for (int r = 0; r < 8; ++r) {
        int grow = bm + wm * 64 + mi * 16 + r + lhalf * 8;
        int gcol = bn + wn * 32 + ni * 16 + lrow;
        if (grow < M && gcol < N) {
          float v = acc[mi][ni][r];
          if (bias)   v += bias[gcol];
          if (addend) v += addend[(size_t)grow * ldc + gcol];
          C[(size_t)grow * ldc + gcol] = v;
        }
      }
}

// ---------------------------------------------------------------------------
// Flash attention, WMMA bf16. Block: 128 threads = 4 waves. Each wave owns
// 16 query rows; 32-key K/V tiles staged (async) in LDS; online softmax.
// Q/K/V: bf16 [B*H, T, 64].  O: bf16 [B*T, 1024] with col = h*64 + e.
// ---------------------------------------------------------------------------
__global__ __launch_bounds__(128) void attn_kernel(
    const unsigned short* __restrict__ Q,
    const unsigned short* __restrict__ Kg,
    const unsigned short* __restrict__ Vg,
    unsigned short* __restrict__ O,
    int T)
{
  __shared__ __align__(16) unsigned short Kt[32 * 64];
  __shared__ __align__(16) unsigned short Vt[32 * 64];
  __shared__ unsigned short Ps[4][16 * 32];

  const int tid = threadIdx.x, lane = tid & 31, wid = tid >> 5;
  const int lrow = lane & 15, lhalf = lane >> 4;
  const int nq = T / 64;
  const int bh = blockIdx.x / nq;
  const int qtile = blockIdx.x % nq;
  const int b = bh >> 4, h = bh & 15;
  const int qbase = qtile * 64 + wid * 16;

  const unsigned short* Qb = Q + (size_t)bh * T * 64;
  const unsigned short* Kb = Kg + (size_t)bh * T * 64;
  const unsigned short* Vb = Vg + (size_t)bh * T * 64;

  // Q A-fragments (rows qbase..qbase+15, two 32-wide k chunks), held in regs
  v16bf aq[2];
  {
    int t  = qbase + lrow;
    int kb = lhalf * 8;
#pragma unroll
    for (int c = 0; c < 2; ++c) {
      V16 tt;
#pragma unroll
      for (int idx = 0; idx < 16; ++idx)
        tt.u[idx] = Qb[(size_t)t * 64 + c * 32 + a_frag_k(idx, kb)];
      aq[c] = tt.v;
    }
  }

  v8f oacc[4] = {};
  float mrun[8], lrun[8];
#pragma unroll
  for (int r = 0; r < 8; ++r) { mrun[r] = -INFINITY; lrun[r] = 0.f; }

  const float scale = 0.125f;   // 1/sqrt(64)

  for (int s0 = 0; s0 < T; s0 += 32) {
    __syncthreads();
    // async-stage K/V tiles: 256 chunks of 16B each, 128 threads
    for (int c = tid; c < 256; c += 128) {
      cp_async_b128(&Kt[c * 8], &Kb[(size_t)s0 * 64 + c * 8]);
      cp_async_b128(&Vt[c * 8], &Vb[(size_t)s0 * 64 + c * 8]);
    }
    cp_async_wait();
    __syncthreads();

    // scores: S[16 x 32] as two 16x16 accumulators
    v8f sacc[2] = {};
#pragma unroll
    for (int nt = 0; nt < 2; ++nt) {
#pragma unroll
      for (int c = 0; c < 2; ++c) {
        V16 bk;
        int key = nt * 16 + lrow;
        int kb  = lhalf * 8;
#pragma unroll
        for (int idx = 0; idx < 16; ++idx)
          bk.u[idx] = Kt[key * 64 + c * 32 + a_frag_k(idx, kb)];   // K^T: B[d][key]
        sacc[nt] = wmma_bf16(aq[c], bk.v, sacc[nt]);
      }
    }

    // online softmax, per-row (row lives in (r, lane-half); 16 lanes = 16 cols)
#pragma unroll
    for (int r = 0; r < 8; ++r) {
      float v0 = sacc[0][r] * scale;
      float v1 = sacc[1][r] * scale;
      float rmax = fmaxf(v0, v1);
      for (int m = 1; m < 16; m <<= 1) rmax = fmaxf(rmax, __shfl_xor(rmax, m));
      float mnew  = fmaxf(mrun[r], rmax);
      float alpha = __expf(mrun[r] - mnew);
      float p0 = __expf(v0 - mnew);
      float p1 = __expf(v1 - mnew);
      float rs = p0 + p1;
      for (int m = 1; m < 16; m <<= 1) rs += __shfl_xor(rs, m);
      lrun[r] = lrun[r] * alpha + rs;
      mrun[r] = mnew;
#pragma unroll
      for (int j = 0; j < 4; ++j) oacc[j][r] *= alpha;
      int row = r + lhalf * 8;
      Ps[wid][row * 32 + lrow]      = f2b(p0);
      Ps[wid][row * 32 + 16 + lrow] = f2b(p1);
    }

    // P as A-fragment (LDS bounce does the lane transpose)
    V16 pa;
    {
      int kb = lhalf * 8;
#pragma unroll
      for (int idx = 0; idx < 16; ++idx)
        pa.u[idx] = Ps[wid][lrow * 32 + a_frag_k(idx, kb)];
    }
    // O += P * V  (4 column tiles over head dim 64)
#pragma unroll
    for (int j = 0; j < 4; ++j) {
      V16 bv;
      int n  = j * 16 + lrow;
      int kb = lhalf * 8;
#pragma unroll
      for (int idx = 0; idx < 16; ++idx)
        bv.u[idx] = Vt[a_frag_k(idx, kb) * 64 + n];
      oacc[j] = wmma_bf16(pa.v, bv.v, oacc[j]);
    }
  }

  // write O (normalized), layout [b*T + t, h*64 + e]
#pragma unroll
  for (int j = 0; j < 4; ++j)
#pragma unroll
    for (int r = 0; r < 8; ++r) {
      int trow = qbase + r + lhalf * 8;
      int col  = h * 64 + j * 16 + lrow;
      float v  = oacc[j][r] / lrun[r];
      O[(size_t)(b * T + trow) * 1024 + col] = f2b(v);
    }
}

// ---------------------------------------------------------------------------
// RMSNorm: one block per row
// ---------------------------------------------------------------------------
__global__ __launch_bounds__(256) void rms_kernel(
    const float* __restrict__ x, const float* __restrict__ w,
    unsigned short* __restrict__ ob, float* __restrict__ of, int D)
{
  __shared__ float red[256];
  int row = blockIdx.x, tid = threadIdx.x;
  const float* xr = x + (size_t)row * D;
  float s = 0.f;
  for (int d = tid; d < D; d += 256) { float v = xr[d]; s += v * v; }
  red[tid] = s; __syncthreads();
  for (int st = 128; st > 0; st >>= 1) {
    if (tid < st) red[tid] += red[tid + st];
    __syncthreads();
  }
  float inv = rsqrtf(red[0] / (float)D + 1e-5f);
  for (int d = tid; d < D; d += 256) {
    float v = xr[d] * inv * w[d];
    ob[(size_t)row * D + d] = f2b(v);
    if (of) of[(size_t)row * D + d] = v;
  }
}

// ---------------------------------------------------------------------------
// Weight converters
// ---------------------------------------------------------------------------
__global__ void conv_bf16_kernel(const float* __restrict__ in,
                                 unsigned short* __restrict__ out, int n)
{
  int i = blockIdx.x * 256 + threadIdx.x;
  if (i < n) out[i] = f2b(in[i]);
}

// row-major convert with padded output leading dim
__global__ void conv_bf16_pad_kernel(const float* __restrict__ in,
                                     unsigned short* __restrict__ out,
                                     int ncols, int ldout, int total)
{
  int i = blockIdx.x * 256 + threadIdx.x;
  if (i >= total) return;
  int r = i / ncols, j = i - r * ncols;
  out[(size_t)r * ldout + j] = f2b(in[i]);
}

// in: [H, Dd, Ep] -> out row-major [Dd, H*Ep]
__global__ void conv_headmajor_kernel(const float* __restrict__ in,
                                      unsigned short* __restrict__ out,
                                      int Dd, int Ep, int H)
{
  int i = blockIdx.x * 256 + threadIdx.x;
  int total = H * Dd * Ep;
  if (i >= total) return;
  int h = i / (Dd * Ep);
  int r = i - h * Dd * Ep;
  int d = r / Ep;
  int e = r - d * Ep;
  out[(size_t)d * (H * Ep) + h * Ep + e] = f2b(in[i]);
}

// ---------------------------------------------------------------------------
// RoPE for Q: qf [N,1024] (bias already added by GEMM) -> qb16 [B*H,T,64]
// ---------------------------------------------------------------------------
__global__ void rope_q_kernel(const float* __restrict__ qf,
                              unsigned short* __restrict__ qo, int T)
{
  int i = blockIdx.x * 256 + threadIdx.x;   // total B*H*T*32
  int ii = i & 31;
  int t  = (i >> 5) % T;
  int bh = i / (T * 32);
  int b = bh >> 4, h = bh & 15;
  size_t row = (size_t)(b * T + t) * 1024 + h * 64;
  float t1 = qf[row + 2 * ii];
  float t2 = qf[row + 2 * ii + 1];
  float invf = __powf(10000.f, -(float)(2 * ii) / 64.f);
  float ang = (float)t * invf;
  float c = __cosf(ang), s = __sinf(ang);
  size_t orow = ((size_t)bh * T + t) * 64;
  qo[orow + ii]      = f2b(t1 * c - t2 * s);
  qo[orow + 32 + ii] = f2b(t1 * s + t2 * c);
}

// ---------------------------------------------------------------------------
// K/V latent expansion: lat [N,512] x bw [16,32,64] -> out bf16 [B*H,T,64]
// ---------------------------------------------------------------------------
__global__ void kv_expand_kernel(const float* __restrict__ lat,
                                 const float* __restrict__ bw,
                                 unsigned short* __restrict__ out,
                                 int T, int do_rope)
{
  int i = blockIdx.x * 256 + threadIdx.x;   // total B*H*T*32
  int ii = i & 31;
  int t  = (i >> 5) % T;
  int bh = i / (T * 32);
  int b = bh >> 4, h = bh & 15;
  const float* lrow = lat + (size_t)(b * T + t) * 512 + h * 32;
  const float* wv = bw + h * 2048;
  float t1 = 0.f, t2 = 0.f;
#pragma unroll 8
  for (int l = 0; l < 32; ++l) {
    float xv = lrow[l];
    t1 += xv * wv[l * 64 + 2 * ii];
    t2 += xv * wv[l * 64 + 2 * ii + 1];
  }
  size_t orow = ((size_t)bh * T + t) * 64;
  if (do_rope) {
    float invf = __powf(10000.f, -(float)(2 * ii) / 64.f);
    float ang = (float)t * invf;
    float c = __cosf(ang), s = __sinf(ang);
    out[orow + ii]      = f2b(t1 * c - t2 * s);
    out[orow + 32 + ii] = f2b(t1 * s + t2 * c);
  } else {
    out[orow + 2 * ii]     = f2b(t1);
    out[orow + 2 * ii + 1] = f2b(t2);
  }
}

// ---------------------------------------------------------------------------
// Router: per-token logits/noise dots, noisy top-2, masked softmax
// ---------------------------------------------------------------------------
__global__ __launch_bounds__(256) void router_kernel(
    const float* __restrict__ xn2,
    const float* __restrict__ rw, const float* __restrict__ rb,
    const float* __restrict__ nw, const float* __restrict__ nb,
    const float* __restrict__ neps,
    float* __restrict__ probs, int* __restrict__ topi, int D)
{
  __shared__ float red[256 * 16];
  int n = blockIdx.x, tid = threadIdx.x;
  const float* xr = xn2 + (size_t)n * D;
  float pr[16];
#pragma unroll
  for (int j = 0; j < 16; ++j) pr[j] = 0.f;
  for (int d = tid; d < D; d += 256) {
    float xv = xr[d];
#pragma unroll
    for (int e = 0; e < 8; ++e) {
      pr[e]     += xv * rw[d * 8 + e];
      pr[8 + e] += xv * nw[d * 8 + e];
    }
  }
#pragma unroll
  for (int j = 0; j < 16; ++j) red[tid * 16 + j] = pr[j];
  __syncthreads();
  for (int st = 128; st > 0; st >>= 1) {
    if (tid < st)
      for (int j = 0; j < 16; ++j)
        red[tid * 16 + j] += red[(tid + st) * 16 + j];
    __syncthreads();
  }
  if (tid == 0) {
    float noisy[8];
    for (int e = 0; e < 8; ++e) {
      float lg = red[e] + rb[e];
      float nr = red[8 + e] + nb[e];
      float sp = (nr > 20.f) ? nr : log1pf(__expf(nr));
      noisy[e] = lg + neps[(size_t)n * 8 + e] * sp;
    }
    int i1 = 0;
    for (int e = 1; e < 8; ++e) if (noisy[e] > noisy[i1]) i1 = e;
    int i2 = -1;
    for (int e = 0; e < 8; ++e)
      if (e != i1 && (i2 < 0 || noisy[e] > noisy[i2])) i2 = e;
    float m  = fmaxf(noisy[i1], noisy[i2]);
    float e1 = __expf(noisy[i1] - m);
    float e2 = __expf(noisy[i2] - m);
    float z  = e1 + e2;
    for (int e = 0; e < 8; ++e) probs[(size_t)n * 8 + e] = 0.f;
    probs[(size_t)n * 8 + i1] = e1 / z;
    probs[(size_t)n * 8 + i2] = e2 / z;
    topi[n * 2]     = i1;
    topi[n * 2 + 1] = i2;
  }
}

// ---------------------------------------------------------------------------
// Aux loss: sum_e (mean_n probs[n,e] - 1/8)^2
// ---------------------------------------------------------------------------
__global__ __launch_bounds__(256) void aux_kernel(
    const float* __restrict__ probs, float* __restrict__ out_aux, int N)
{
  __shared__ float red[256 * 8];
  int tid = threadIdx.x;
  float s[8];
#pragma unroll
  for (int e = 0; e < 8; ++e) s[e] = 0.f;
  for (int n = tid; n < N; n += 256)
#pragma unroll
    for (int e = 0; e < 8; ++e) s[e] += probs[(size_t)n * 8 + e];
#pragma unroll
  for (int e = 0; e < 8; ++e) red[tid * 8 + e] = s[e];
  __syncthreads();
  for (int st = 128; st > 0; st >>= 1) {
    if (tid < st)
      for (int e = 0; e < 8; ++e)
        red[tid * 8 + e] += red[(tid + st) * 8 + e];
    __syncthreads();
  }
  if (tid == 0) {
    float aux = 0.f;
    for (int e = 0; e < 8; ++e) {
      float d = red[e] / (float)N - 0.125f;
      aux += d * d;
    }
    *out_aux = aux;
  }
}

// ---------------------------------------------------------------------------
// Capacity dispatch (in-order cumsum semantics): 1 block, 8 threads
// ---------------------------------------------------------------------------
__global__ void capacity_kernel(const int* __restrict__ topi,
                                const float* __restrict__ probs,
                                int* __restrict__ tok, float* __restrict__ gate,
                                int N, int cap)
{
  int e = threadIdx.x;
  if (e >= 8) return;
  for (int i = 0; i < cap; ++i) { tok[e * cap + i] = 0; gate[e * cap + i] = 0.f; }
  int cnt = 0;
  for (int n = 0; n < N; ++n) {
    if (topi[n * 2] == e || topi[n * 2 + 1] == e) {
      if (cnt < cap) {
        tok[e * cap + cnt]  = n;
        gate[e * cap + cnt] = probs[(size_t)n * 8 + e];
      }
      cnt++;
    }
  }
}

__global__ void gather_kernel(const unsigned short* __restrict__ xn2b,
                              const int* __restrict__ tok_e,
                              unsigned short* __restrict__ xe, int D, int cap)
{
  int i = blockIdx.x * 256 + threadIdx.x;
  if (i >= cap * D) return;
  int s = i / D, d = i - s * D;
  xe[i] = xn2b[(size_t)tok_e[s] * D + d];
}

__global__ void swiglu_kernel(const float* __restrict__ hbuf,
                              unsigned short* __restrict__ hc,
                              int cap, int HIDc, int ldhc)
{
  int i = blockIdx.x * 256 + threadIdx.x;
  if (i >= cap * HIDc) return;
  int row = i / HIDc, j = i - row * HIDc;
  float h1 = hbuf[(size_t)row * (2 * HIDc) + j];
  float h2 = hbuf[(size_t)row * (2 * HIDc) + HIDc + j];
  float sig = 1.f / (1.f + __expf(-h1));
  hc[(size_t)row * ldhc + j] = f2b(h1 * sig * h2);
}

__global__ void scatter_kernel(const float* __restrict__ oe,
                               const int* __restrict__ tok_e,
                               const float* __restrict__ gate_e,
                               float* __restrict__ upd, int D, int cap)
{
  int i = blockIdx.x * 256 + threadIdx.x;
  if (i >= cap * D) return;
  int s = i / D, d = i - s * D;
  float g = gate_e[s];
  if (g != 0.f) atomicAdd(&upd[(size_t)tok_e[s] * D + d], oe[i] * g);
}

__global__ void addout_kernel(const float* __restrict__ x2,
                              const float* __restrict__ upd,
                              float* __restrict__ out, int n)
{
  int i = blockIdx.x * 256 + threadIdx.x;
  if (i < n) out[i] = x2[i] + upd[i];
}

// ---------------------------------------------------------------------------
// Host
// ---------------------------------------------------------------------------
static void launch_gemm(const unsigned short* A, int lda,
                        const unsigned short* Bm, int ldb,
                        float* C, int ldc, int M, int Nn, int K,
                        const float* bias, const float* add, hipStream_t s)
{
  dim3 g((Nn + BN - 1) / BN, (M + BM - 1) / BM);
  gemm_bf16_kernel<<<g, 256, 0, s>>>(A, lda, Bm, ldb, C, ldc, M, Nn, K, bias, add);
}

extern "C" void kernel_launch(void* const* d_in, const int* in_sizes, int n_in,
                              void* d_out, int out_size, void* d_ws, size_t ws_size,
                              hipStream_t stream)
{
  (void)in_sizes; (void)n_in; (void)out_size; (void)ws_size;
  constexpr int Bb = 4, T = 1024, D = 1024, H = 16;
  constexpr int N = Bb * T;                 // 4096 tokens
  constexpr int HID = 2730, TWOHID = 5460;
  constexpr int SWI_LD = 5472;              // padded ld (16B-aligned rows)
  constexpr int HID_LD = 2736;              // padded ld (16B-aligned rows)
  constexpr int CAP = 1024;                 // N*TOP_K/E * 1.0

  const float* x       = (const float*)d_in[0];
  const float* neps    = (const float*)d_in[1];
  const float* ln1_w   = (const float*)d_in[2];
  const float* ln2_w   = (const float*)d_in[3];
  const float* q_w     = (const float*)d_in[4];
  const float* q_b     = (const float*)d_in[5];
  const float* ka_w    = (const float*)d_in[6];
  const float* kb_w    = (const float*)d_in[7];
  const float* va_w    = (const float*)d_in[8];
  const float* vb_w    = (const float*)d_in[9];
  const float* proj_w  = (const float*)d_in[10];
  const float* proj_b  = (const float*)d_in[11];
  const float* route_w = (const float*)d_in[12];
  const float* route_b = (const float*)d_in[13];
  const float* noise_w = (const float*)d_in[14];
  const float* noise_b = (const float*)d_in[15];
  const float* swi_w   = (const float*)d_in[16];
  const float* down_w  = (const float*)d_in[17];

  char* ws = (char*)d_ws;
  size_t off = 0;
  auto alloc = [&](size_t bytes) -> void* {
    void* p = ws + off;
    off = (off + bytes + 255) & ~(size_t)255;
    return p;
  };

  unsigned short* qwb   = (unsigned short*)alloc((size_t)1024 * 1024 * 2);
  unsigned short* kawb  = (unsigned short*)alloc((size_t)1024 * 512 * 2);
  unsigned short* vawb  = (unsigned short*)alloc((size_t)1024 * 512 * 2);
  unsigned short* pjwb  = (unsigned short*)alloc((size_t)1024 * 1024 * 2);
  unsigned short* swib  = (unsigned short*)alloc((size_t)8 * 1024 * SWI_LD * 2);
  unsigned short* dwnb  = (unsigned short*)alloc((size_t)8 * HID * 1024 * 2);
  unsigned short* xn1b  = (unsigned short*)alloc((size_t)N * D * 2);
  float*          qf    = (float*)alloc((size_t)N * 1024 * 4);
  float*          klf   = (float*)alloc((size_t)N * 512 * 4);
  float*          vlf   = (float*)alloc((size_t)N * 512 * 4);
  unsigned short* qb16  = (unsigned short*)alloc((size_t)N * 1024 * 2);
  unsigned short* kb16  = (unsigned short*)alloc((size_t)N * 1024 * 2);
  unsigned short* vb16  = (unsigned short*)alloc((size_t)N * 1024 * 2);
  unsigned short* ob16  = (unsigned short*)alloc((size_t)N * 1024 * 2);
  float*          x2f   = (float*)alloc((size_t)N * D * 4);
  unsigned short* xn2b  = (unsigned short*)alloc((size_t)N * D * 2);
  float*          xn2f  = (float*)alloc((size_t)N * D * 4);
  float*          probs = (float*)alloc((size_t)N * 8 * 4);
  int*            topi  = (int*)alloc((size_t)N * 2 * 4);
  int*            tok   = (int*)alloc((size_t)8 * CAP * 4);
  float*          gate  = (float*)alloc((size_t)8 * CAP * 4);
  unsigned short* xeb   = (unsigned short*)alloc((size_t)CAP * D * 2);
  float*          hf    = (float*)alloc((size_t)CAP * TWOHID * 4);
  unsigned short* hcb   = (unsigned short*)alloc((size_t)CAP * HID_LD * 2);
  float*          oef   = (float*)alloc((size_t)CAP * D * 4);
  float*          updf  = (float*)alloc((size_t)N * D * 4);

  (void)hipMemsetAsync(updf, 0, (size_t)N * D * 4, stream);

  // --- weight conversion to bf16 -------------------------------------------
  {
    int n;
    n = H * 1024 * 64;
    conv_headmajor_kernel<<<(n + 255) / 256, 256, 0, stream>>>(q_w, qwb, 1024, 64, H);
    n = H * 1024 * 32;
    conv_headmajor_kernel<<<(n + 255) / 256, 256, 0, stream>>>(ka_w, kawb, 1024, 32, H);
    conv_headmajor_kernel<<<(n + 255) / 256, 256, 0, stream>>>(va_w, vawb, 1024, 32, H);
    n = 1024 * 1024;
    conv_bf16_kernel<<<(n + 255) / 256, 256, 0, stream>>>(proj_w, pjwb, n);
    n = 8 * 1024 * TWOHID;   // padded rows: [8*1024, 5460] -> ld 5472
    conv_bf16_pad_kernel<<<(n + 255) / 256, 256, 0, stream>>>(swi_w, swib,
                                                              TWOHID, SWI_LD, n);
    n = 8 * HID * 1024;
    conv_bf16_kernel<<<(n + 255) / 256, 256, 0, stream>>>(down_w, dwnb, n);
  }

  // --- attention path -------------------------------------------------------
  rms_kernel<<<N, 256, 0, stream>>>(x, ln1_w, xn1b, nullptr, D);

  launch_gemm(xn1b, 1024, qwb, 1024, qf, 1024, N, 1024, 1024, q_b, nullptr, stream);
  launch_gemm(xn1b, 1024, kawb, 512, klf, 512, N, 512, 1024, nullptr, nullptr, stream);
  launch_gemm(xn1b, 1024, vawb, 512, vlf, 512, N, 512, 1024, nullptr, nullptr, stream);

  {
    int n = Bb * H * T * 32;   // 2,097,152
    rope_q_kernel<<<(n + 255) / 256, 256, 0, stream>>>(qf, qb16, T);
    kv_expand_kernel<<<(n + 255) / 256, 256, 0, stream>>>(klf, kb_w, kb16, T, 1);
    kv_expand_kernel<<<(n + 255) / 256, 256, 0, stream>>>(vlf, vb_w, vb16, T, 0);
  }

  attn_kernel<<<Bb * H * (T / 64), 128, 0, stream>>>(qb16, kb16, vb16, ob16, T);

  // x2 = x + o @ proj_w + proj_b
  launch_gemm(ob16, 1024, pjwb, 1024, x2f, 1024, N, 1024, 1024, proj_b, x, stream);

  // --- MoE path -------------------------------------------------------------
  rms_kernel<<<N, 256, 0, stream>>>(x2f, ln2_w, xn2b, xn2f, D);

  router_kernel<<<N, 256, 0, stream>>>(xn2f, route_w, route_b, noise_w, noise_b,
                                       neps, probs, topi, D);
  aux_kernel<<<1, 256, 0, stream>>>(probs, (float*)d_out + (size_t)N * D, N);
  capacity_kernel<<<1, 8, 0, stream>>>(topi, probs, tok, gate, N, CAP);

  for (int e = 0; e < 8; ++e) {
    const int* tok_e = tok + e * CAP;
    const float* gate_e = gate + e * CAP;
    int n = CAP * D;
    gather_kernel<<<(n + 255) / 256, 256, 0, stream>>>(xn2b, tok_e, xeb, D, CAP);
    launch_gemm(xeb, 1024, swib + (size_t)e * 1024 * SWI_LD, SWI_LD,
                hf, TWOHID, CAP, TWOHID, 1024, nullptr, nullptr, stream);
    int ns = CAP * HID;
    swiglu_kernel<<<(ns + 255) / 256, 256, 0, stream>>>(hf, hcb, CAP, HID, HID_LD);
    launch_gemm(hcb, HID_LD, dwnb + (size_t)e * HID * 1024, 1024,
                oef, 1024, CAP, 1024, HID, nullptr, nullptr, stream);
    scatter_kernel<<<(n + 255) / 256, 256, 0, stream>>>(oef, tok_e, gate_e, updf, D, CAP);
  }

  {
    int n = N * D;
    addout_kernel<<<(n + 255) / 256, 256, 0, stream>>>(x2f, updf, (float*)d_out, n);
  }
}
